// GaussianActor_72825465471311
// MI455X (gfx1250) — matrix operand order repaired and checked by reading, not compiled, not run on hardware
//
#include <hip/hip_runtime.h>
#include <hip/hip_bf16.h>
#include <math.h>

// ---------------------------------------------------------------------------
// GaussianActor forward for MI455X (gfx1250, wave32, WMMA + TDM).
//   trunk:  h1 = relu(states @ W1 + b1)   [B,376]x[376,256]
//           h2 = relu(h1 @ W2 + b2)       [B,256]x[256,256]
//   heads:  mu = h2 @ Wmu + bmu, ls = clip(h2 @ Wsig + bsig, -20, 2)
//   out:    actions = tanh(mu + exp(ls)*noise), log_pis
// GEMMs on v_wmma_f32_16x16x32_f16 (f32 accum). Weights pre-converted to f16
// in d_ws and DMA'd into LDS per K-chunk with tensor_load_to_lds (TENSORcnt).
// States staged via vectorized float4 loads -> packed f16 -> ds_store_b128.
// ---------------------------------------------------------------------------

typedef __attribute__((ext_vector_type(16))) _Float16 v16h;
typedef __attribute__((ext_vector_type(8)))  _Float16 v8h;
typedef __attribute__((ext_vector_type(8)))  float    v8f;
typedef __attribute__((ext_vector_type(4)))  unsigned int v4u;
typedef __attribute__((ext_vector_type(8)))  int      v8i;
typedef __attribute__((ext_vector_type(4)))  int      v4i;

#define SD   376   // state dim (376 % 8 == 0 -> uniform 8-wide staging groups)
#define HDIM 256   // hidden width
#define AD   17    // action dim
#define TM   64    // batch rows per workgroup
#define KC   32    // K-chunk (== WMMA K for f16)
#define KP1  384   // layer-1 K padded to multiple of 32

#define WH1_OFF 0                    // f16 W1 padded [384 x 256] in d_ws
#define WH2_OFF (384 * 256)          // f16 W2        [256 x 256]
#define WHH_OFF (WH2_OFF + 256*256)  // f16 heads     [256 x 64] (mu|sig, padded)
#define WH_TOTAL (WHH_OFF + 256*64)  // total f16 elements in d_ws

#define LOG2PI_F 1.8378770664093453f

// ---- WMMA fragment loaders (ISA 7.12.2 wave32 layouts) --------------------
__device__ __forceinline__ v16h load_a_frag(const _Float16* base, int rowStride,
                                            int mBase, int kBase, int lane) {
  const int m    = mBase + (lane & 15);
  const int kOff = kBase + ((lane >> 4) << 3);
  const _Float16* p = base + m * rowStride + kOff;
  v8h lo = *(const v8h*)(p);
  v8h hi = *(const v8h*)(p + 16);
  v16h a;
#pragma unroll
  for (int i = 0; i < 8; ++i) { a[i] = lo[i]; a[i + 8] = hi[i]; }
  return a;
}

__device__ __forceinline__ v16h load_b_frag(const _Float16* base, int rowStride,
                                            int nBase, int lane) {
  const int n    = nBase + (lane & 15);
  const int kOff = (lane >> 4) << 3;
  v16h b;
#pragma unroll
  for (int i = 0; i < 8; ++i) {
    b[i]     = base[(kOff + i) * rowStride + n];
    b[i + 8] = base[(kOff + 16 + i) * rowStride + n];
  }
  return b;
}

__device__ __forceinline__ v8f wmma_f16(v16h a, v16h b, v8f c) {
  return __builtin_amdgcn_wmma_f32_16x16x32_f16(false, a, false, b,
                                                (short)0, c, false, false);
}

// ---- TDM: 2D tile (rows x cols f16, row-major, fully in-bounds) -> LDS ----
// D# per CDNA5 ISA 8.3/8.4: group0 {count=1, lds_addr, global_addr, type=2},
// group1 {data_size=2B, tensor_dim0=cols, tensor_dim1=rows, tile=cols x rows,
//         dim0_stride=cols}. Groups 2/3 zero (2D tensor).
__device__ __forceinline__ void tdm_load_2d(unsigned lds_off, const _Float16* src,
                                            unsigned cols, unsigned rows) {
  const unsigned long long ga = (unsigned long long)(uintptr_t)src;
  v4u g0;
  g0[0] = 1u;                                   // count=1, user descriptor
  g0[1] = lds_off;                              // lds_addr (bytes)
  g0[2] = (unsigned)ga;                         // global_addr[31:0]
  g0[3] = (unsigned)((ga >> 32) & 0x01FFFFFFu)  // global_addr[56:32]
          | (2u << 30);                         // type = 2 ("image")
  v8i g1;
  g1[0] = (int)(1u << 16);                      // data_size = 1 -> 2 bytes
  g1[1] = 0;                                    // no atomic barrier
  g1[2] = (int)((cols & 0xFFFFu) << 16);        // tensor_dim0[15:0] @ bits 63:48
  g1[3] = (int)(((cols >> 16) & 0xFFFFu)        // tensor_dim0[31:16]
          | ((rows & 0xFFFFu) << 16));          // tensor_dim1[15:0]... rows<64K
  // tile_dim0 @ bits 127:112 lives in g1[3] high half? No: bits 112..127 are
  // word3[31:16]; tensor_dim1[15:0] occupies word2[31:16]. Recompute exactly:
  g1[2] = (int)(((cols & 0xFFFFu) << 16));                       // w1: dim0 lo
  // w2 = bits 95:64 : dim0 hi (15:0) | dim1 lo (31:16)
  g1[3] = (int)(((cols >> 16) & 0xFFFFu) | ((rows & 0xFFFFu) << 16));
  g1[4] = (int)(((rows >> 16) & 0xFFFFu) | ((cols & 0xFFFFu) << 16)); // w3: dim1 hi | tile_dim0
  g1[5] = (int)(rows & 0xFFFFu);                // w4: tile_dim1 | tile_dim2=0
  g1[6] = (int)cols;                            // w5: tensor_dim0_stride[31:0]
  g1[7] = 0;                                    // w6: stride hi16 | dim1_stride lo
  // NOTE: v8i only has 8 words; group1 is 256 bits = 8 words. w7 (dim1_stride
  // hi) would be g1[7]'s neighbor -- layout above packs w0..w7 as g1[0..7].
  // Shift all down by one: rebuild properly below.
  v8i G1;
  G1[0] = (int)(1u << 16);                                            // w0
  G1[1] = (int)((cols & 0xFFFFu) << 16);                              // w1
  G1[2] = (int)(((cols >> 16) & 0xFFFFu) | ((rows & 0xFFFFu) << 16)); // w2
  G1[3] = (int)(((rows >> 16) & 0xFFFFu) | ((cols & 0xFFFFu) << 16)); // w3
  G1[4] = (int)(rows & 0xFFFFu);                                      // w4
  G1[5] = (int)cols;                                                  // w5
  G1[6] = 0;                                                          // w6
  G1[7] = 0;                                                          // w7
  v4i z4 = {};
#if defined(__clang_major__) && (__clang_major__ >= 23)
  v8i z8 = {};
  __builtin_amdgcn_tensor_load_to_lds(g0, G1, z4, z4, z8, 0);
#else
  __builtin_amdgcn_tensor_load_to_lds(g0, G1, z4, z4, 0);
#endif
}

// ---- weight pre-conversion: f32 -> padded f16 images in d_ws --------------
__global__ void __launch_bounds__(256)
convert_weights_kernel(const float* __restrict__ W1, const float* __restrict__ W2,
                       const float* __restrict__ Wmu, const float* __restrict__ Wsg,
                       _Float16* __restrict__ wh) {
  const int i = blockIdx.x * 256 + threadIdx.x;
  if (i < WH2_OFF) {                       // W1 padded to 384 rows
    const int k = i >> 8, n = i & 255;
    wh[WH1_OFF + i] = (_Float16)((k < SD) ? W1[(size_t)k * HDIM + n] : 0.0f);
  } else if (i < WHH_OFF) {                // W2
    const int j = i - WH2_OFF;
    wh[i] = (_Float16)W2[j];
  } else if (i < WH_TOTAL) {               // heads: cols 0..31 mu, 32..63 sig
    const int j = i - WHH_OFF;
    const int k = j >> 6, c = j & 63;
    float v;
    if (c < 32) v = (c < AD) ? Wmu[(size_t)k * AD + c] : 0.0f;
    else { const int n2 = c - 32; v = (n2 < AD) ? Wsg[(size_t)k * AD + n2] : 0.0f; }
    wh[i] = (_Float16)v;
  }
}

// ---- main fused actor kernel ----------------------------------------------
template <bool USE_TDM>
__global__ void __launch_bounds__(256)
gaussian_actor_kernel(const float* __restrict__ states,
                      const float* __restrict__ W1,  const float* __restrict__ b1,
                      const float* __restrict__ W2,  const float* __restrict__ b2,
                      const float* __restrict__ Wmu, const float* __restrict__ bmu,
                      const float* __restrict__ Wsg, const float* __restrict__ bsg,
                      const float* __restrict__ noise,
                      const _Float16* __restrict__ wh,
                      float* __restrict__ actions, float* __restrict__ logpis) {
  __shared__ _Float16 sH[TM * HDIM];        // 32 KB: h1 then h2 (f16)
  __shared__ _Float16 sW[KC * HDIM];        // 16 KB: weight K-chunk staging
  __shared__ _Float16 sA[TM * KC];          //  4 KB: states K-chunk (f16)
  __shared__ float    sHead[TM * (2 * AD)]; // per-row mu[0..16], ls[17..33]

  const int tid   = threadIdx.x;
  const int lane  = tid & 31;
  const int w     = tid >> 5;          // wave id 0..7
  const int mBase = (w & 3) << 4;      // 4 row-blocks of 16
  const int nHalf = (w >> 2) << 7;     // N in [0,128) or [128,256)
  const int row0  = blockIdx.x * TM;
  const unsigned sW_off = (unsigned)(uintptr_t)(void*)sW;

  v8f acc[8];
  const v8f vzero = {};

  // ---------------- Layer 1: h1 = relu(states @ W1 + b1) ------------------
#pragma unroll
  for (int t = 0; t < 8; ++t) acc[t] = vzero;

  for (int kc = 0; kc < KP1 / KC; ++kc) {
    __syncthreads();  // previous chunk's LDS reads finished
    if (USE_TDM) {
      if (w == 0)  // one DMA per block: f16 W1 chunk [32 x 256] -> sW
        tdm_load_2d(sW_off, wh + WH1_OFF + (size_t)kc * KC * HDIM, HDIM, KC);
    } else {
      // vectorized VGPR staging: thread -> one row of chunk, 32 halves
      const int kr = kc * KC + (tid >> 3);
      const int n0 = (tid & 7) << 5;
#pragma unroll
      for (int g = 0; g < 4; ++g) {
        v8h hv;
        if (kr < SD) {
          const float4 f0 = *(const float4*)&W1[(size_t)kr * HDIM + n0 + g * 8];
          const float4 f1 = *(const float4*)&W1[(size_t)kr * HDIM + n0 + g * 8 + 4];
          hv[0] = (_Float16)f0.x; hv[1] = (_Float16)f0.y;
          hv[2] = (_Float16)f0.z; hv[3] = (_Float16)f0.w;
          hv[4] = (_Float16)f1.x; hv[5] = (_Float16)f1.y;
          hv[6] = (_Float16)f1.z; hv[7] = (_Float16)f1.w;
        } else {
          hv = (v8h){};
        }
        *(v8h*)&sW[(tid >> 3) * HDIM + n0 + g * 8] = hv;
      }
    }
    // states chunk [64 x 32]: thread -> (row, 8-col group); 376 % 8 == 0 so
    // the K<376 predicate is uniform per group (no per-lane masking).
    {
      const int r  = tid >> 2;
      const int j0 = (tid & 3) << 3;
      const int k0 = kc * KC + j0;
      v8h hv;
      if (k0 < SD) {
        const float* sp = &states[(size_t)(row0 + r) * SD + k0];
        const float4 f0 = *(const float4*)(sp);
        const float4 f1 = *(const float4*)(sp + 4);
        hv[0] = (_Float16)f0.x; hv[1] = (_Float16)f0.y;
        hv[2] = (_Float16)f0.z; hv[3] = (_Float16)f0.w;
        hv[4] = (_Float16)f1.x; hv[5] = (_Float16)f1.y;
        hv[6] = (_Float16)f1.z; hv[7] = (_Float16)f1.w;
      } else {
        hv = (v8h){};
      }
      *(v8h*)&sA[r * KC + j0] = hv;
    }
    if (USE_TDM) {
      if (w == 0) __builtin_amdgcn_s_wait_tensorcnt(0);
    }
    __syncthreads();
    const v16h a = load_a_frag(sA, KC, mBase, 0, lane);
#pragma unroll
    for (int t = 0; t < 8; ++t) {
      const v16h bm = load_b_frag(sW, HDIM, nHalf + (t << 4), lane);
      acc[t] = wmma_f16(a, bm, acc[t]);
    }
  }
  // epilogue: bias + relu -> sH (f16)
#pragma unroll
  for (int t = 0; t < 8; ++t) {
    const int n = nHalf + (t << 4) + (lane & 15);
    const float bias = b1[n];
#pragma unroll
    for (int v = 0; v < 8; ++v) {
      const int m = mBase + v + ((lane >> 4) << 3);
      const float h = acc[t][v] + bias;
      sH[m * HDIM + n] = (_Float16)(h > 0.0f ? h : 0.0f);
    }
  }
  __syncthreads();

  // ---------------- Layer 2: h2 = relu(h1 @ W2 + b2) ----------------------
#pragma unroll
  for (int t = 0; t < 8; ++t) acc[t] = vzero;

  for (int kc = 0; kc < HDIM / KC; ++kc) {
    __syncthreads();
    if (USE_TDM) {
      if (w == 0) {
        tdm_load_2d(sW_off, wh + WH2_OFF + (size_t)kc * KC * HDIM, HDIM, KC);
        __builtin_amdgcn_s_wait_tensorcnt(0);
      }
    } else {
      const int kr = kc * KC + (tid >> 3);
      const int n0 = (tid & 7) << 5;
#pragma unroll
      for (int g = 0; g < 4; ++g) {
        const float4 f0 = *(const float4*)&W2[(size_t)kr * HDIM + n0 + g * 8];
        const float4 f1 = *(const float4*)&W2[(size_t)kr * HDIM + n0 + g * 8 + 4];
        v8h hv;
        hv[0] = (_Float16)f0.x; hv[1] = (_Float16)f0.y;
        hv[2] = (_Float16)f0.z; hv[3] = (_Float16)f0.w;
        hv[4] = (_Float16)f1.x; hv[5] = (_Float16)f1.y;
        hv[6] = (_Float16)f1.z; hv[7] = (_Float16)f1.w;
        *(v8h*)&sW[(tid >> 3) * HDIM + n0 + g * 8] = hv;
      }
    }
    __syncthreads();
    const v16h a = load_a_frag(sH, HDIM, mBase, kc * KC, lane);
#pragma unroll
    for (int t = 0; t < 8; ++t) {
      const v16h bm = load_b_frag(sW, HDIM, nHalf + (t << 4), lane);
      acc[t] = wmma_f16(a, bm, acc[t]);
    }
  }
  __syncthreads();  // everyone done reading h1 before overwrite
#pragma unroll
  for (int t = 0; t < 8; ++t) {
    const int n = nHalf + (t << 4) + (lane & 15);
    const float bias = b2[n];
#pragma unroll
    for (int v = 0; v < 8; ++v) {
      const int m = mBase + v + ((lane >> 4) << 3);
      const float h = acc[t][v] + bias;
      sH[m * HDIM + n] = (_Float16)(h > 0.0f ? h : 0.0f);
    }
  }
  __syncthreads();

  // ---------------- Heads: mu / log_sigma via WMMA (N padded 17->32) ------
  // waves 0..3 -> mu (nB0 = 0), waves 4..7 -> log_sigma (nB0 = 32)
  acc[0] = vzero;
  acc[1] = vzero;
  const int nB0 = (w >> 2) << 5;
  for (int kc = 0; kc < HDIM / KC; ++kc) {
    __syncthreads();
    if (USE_TDM) {
      if (w == 0) {
        tdm_load_2d(sW_off, wh + WHH_OFF + (size_t)kc * KC * 64, 64, KC);
        __builtin_amdgcn_s_wait_tensorcnt(0);
      }
    } else {
      // stage [32 x 64]: cols 0..31 = Wmu (pad), cols 32..63 = Wsig (pad)
      for (int idx = tid; idx < KC * 64; idx += 256) {
        const int i = idx >> 6, c = idx & 63;
        const int k = kc * KC + i;
        float v;
        if (c < 32) v = (c < AD) ? Wmu[(size_t)k * AD + c] : 0.0f;
        else { const int n2 = c - 32; v = (n2 < AD) ? Wsg[(size_t)k * AD + n2] : 0.0f; }
        sW[idx] = (_Float16)v;
      }
    }
    __syncthreads();
    const v16h a = load_a_frag(sH, HDIM, mBase, kc * KC, lane);
#pragma unroll
    for (int t = 0; t < 2; ++t) {
      const v16h bm = load_b_frag(sW, 64, nB0 + (t << 4), lane);
      acc[t] = wmma_f16(a, bm, acc[t]);
    }
  }
  // scatter head results (+bias) to sHead
#pragma unroll
  for (int t = 0; t < 2; ++t) {
    const int nGlob = nB0 + (t << 4) + (lane & 15);  // 0..63
    const int col   = nGlob & 31;
    const int isSig = nGlob >> 5;
    if (col < AD) {
      const float bias = isSig ? bsg[col] : bmu[col];
#pragma unroll
      for (int v = 0; v < 8; ++v) {
        const int m = mBase + v + ((lane >> 4) << 3);
        sHead[m * (2 * AD) + isSig * AD + col] = acc[t][v] + bias;
      }
    }
  }
  __syncthreads();

  // ---------------- Sampling epilogue: one thread per batch row -----------
  if (tid < TM) {
    const int r = tid;
    const size_t R = (size_t)row0 + r;
    float s_n2 = 0.0f, s_ls = 0.0f, s_lg = 0.0f;
#pragma unroll 1
    for (int a = 0; a < AD; ++a) {
      const float mu = sHead[r * (2 * AD) + a];
      float ls = sHead[r * (2 * AD) + AD + a];
      ls = fminf(2.0f, fmaxf(-20.0f, ls));
      const float sg  = expf(ls);
      const float nz  = noise[R * AD + a];
      const float raw = mu + sg * nz;           // sample (stop-gradient)
      const float th  = tanhf(raw);
      actions[R * AD + a] = th;                 // MAX_ACTION == 1.0
      s_n2 += nz * nz;                          // z = (raw-mu)/sigma == noise
      s_ls += ls;
      s_lg += logf(1.0f - th * th + 1e-6f);
    }
    logpis[R] = -0.5f * s_n2 - s_ls - 0.5f * (float)AD * LOG2PI_F - s_lg;
  }
}

extern "C" void kernel_launch(void* const* d_in, const int* in_sizes, int n_in,
                              void* d_out, int out_size, void* d_ws, size_t ws_size,
                              hipStream_t stream) {
  (void)n_in; (void)out_size;
  const float* states = (const float*)d_in[0];
  const float* W1     = (const float*)d_in[1];
  const float* b1     = (const float*)d_in[2];
  const float* W2     = (const float*)d_in[3];
  const float* b2     = (const float*)d_in[4];
  const float* Wmu    = (const float*)d_in[5];
  const float* bmu    = (const float*)d_in[6];
  const float* Wsg    = (const float*)d_in[7];
  const float* bsg    = (const float*)d_in[8];
  const float* noise  = (const float*)d_in[9];

  const int Bn = in_sizes[0] / SD;           // 262144
  float* actions = (float*)d_out;            // [B, 17] flat
  float* logpis  = actions + (size_t)Bn * AD;

  dim3 grid(Bn / TM);                        // 4096 blocks (B % 64 == 0)
  dim3 block(256);                           // 8 wave32 per workgroup

  const size_t need = (size_t)WH_TOTAL * sizeof(_Float16);  // ~352 KB
  if (ws_size >= need && d_ws != nullptr) {
    _Float16* wh = (_Float16*)d_ws;
    convert_weights_kernel<<<dim3((WH_TOTAL + 255) / 256), block, 0, stream>>>(
        W1, W2, Wmu, Wsg, wh);
    gaussian_actor_kernel<true><<<grid, block, 0, stream>>>(
        states, W1, b1, W2, b2, Wmu, bmu, Wsg, bsg, noise, wh, actions, logpis);
  } else {
    gaussian_actor_kernel<false><<<grid, block, 0, stream>>>(
        states, W1, b1, W2, b2, Wmu, bmu, Wsg, bsg, noise, nullptr, actions, logpis);
  }
}
